// HeteroGAE_geo_Decoder_pairwise_3985729650716
// MI455X (gfx1250) — matrix operand order repaired and checked by reading, not compiled, not run on hardware
//
#include <hip/hip_runtime.h>
#include <hip/hip_bf16.h>
#include <cstdint>

// ---------------------------------------------------------------------------
// CDNA5 WMMA types / helpers (gfx1250, wave32)
// ---------------------------------------------------------------------------
typedef __attribute__((ext_vector_type(16))) _Float16 v16h;
typedef __attribute__((ext_vector_type(8)))  float    v8f;

union Frag16 { v16h h; uint32_t u[8]; };

__device__ __forceinline__ float gelu_f(float x) {
    // exact erf GELU (torch.nn.GELU default)
    return 0.5f * x * (1.0f + erff(x * 0.70710678118654752f));
}

// A-fragment (16x32 f16) from an LDS tile stored row-major [row][k] halfs.
// ISA layout: lanes 0-15 row=lane, halfs = K{0,1},{2,3},{4,5},{6,7},{16,17}..{22,23}
//             lanes 16-31 same rows, K offset +8.  => dword d covers K=2d,2d+1.
__device__ __forceinline__ v16h load_a_lds(const _Float16* base, int ldh, int kbase, int lane) {
    int r = lane & 15;
    int g = (lane >> 4) & 1;
    const uint32_t* q = reinterpret_cast<const uint32_t*>(base + r * ldh + kbase);
    Frag16 f;
    int b0 = g * 4;
#pragma unroll
    for (int j = 0; j < 4; ++j) { f.u[j] = q[b0 + j]; f.u[4 + j] = q[8 + b0 + j]; }
    return f.h;
}

// B-fragment (32x16 f16) from weights pre-packed as:
//   Wp[(((kc*Nout + col)*2 + laneGroup)*16 + h]  with K = kc*32 + (h>>3)*16 + laneGroup*8 + (h&7)
// so each lane reads 16 contiguous halfs (two b128 loads).
__device__ __forceinline__ v16h load_b_packed(const _Float16* Wp, int Nout, int kc, int colbase, int lane) {
    int col = colbase + (lane & 15);
    int g = (lane >> 4) & 1;
    const uint32_t* q = reinterpret_cast<const uint32_t*>(Wp + ((size_t)(kc * Nout + col) * 2 + g) * 16);
    Frag16 f;
#pragma unroll
    for (int j = 0; j < 8; ++j) f.u[j] = q[j];
    return f.h;
}

__device__ __forceinline__ v8f wmma16(v16h a, v16h b, v8f c) {
    return __builtin_amdgcn_wmma_f32_16x16x32_f16(false, a, false, b, (short)0, c, false, false);
}

// ---------------------------------------------------------------------------
// Utility kernels
// ---------------------------------------------------------------------------
__global__ void k_fill_zero(float* p, int n) {
    int i = blockIdx.x * 256 + threadIdx.x;
    if (i < n) p[i] = 0.0f;
}

// Pack fp32 row-major weight [K][Nout] into WMMA B-fragment-friendly f16 layout.
__global__ void k_pack_b(const float* __restrict__ W, _Float16* __restrict__ Wp, int K, int Nout) {
    int t = blockIdx.x * 256 + threadIdx.x;
    if (t >= K * Nout) return;
    int k = t / Nout, col = t - k * Nout;
    int kc = k >> 5, r = k & 31, g = (r >> 3) & 1, h = (r & 7) | ((r >> 4) << 3);
    Wp[((size_t)(kc * Nout + col) * 2 + g) * 16 + h] = (_Float16)W[(size_t)k * Nout + col];
}

// Edge scatter: acc[dst] += x[src]; cnt[dst] += 1.  One thread per (edge, 4 feats).
__global__ void k_scatter(const float* __restrict__ x, int ldx,
                          const int* __restrict__ src, const int* __restrict__ dst,
                          float* __restrict__ acc, float* __restrict__ cnt, int E) {
    long long t = (long long)blockIdx.x * 256 + threadIdx.x;
    if (t >= (long long)E * 32) return;
    int e = (int)(t >> 5);
    int c = ((int)t & 31) * 4;
    int s = src[e], d = dst[e];
    float4 v = *(const float4*)(x + (size_t)s * ldx + c);
    float* a = acc + (size_t)d * 128 + c;
    atomicAdd(a + 0, v.x); atomicAdd(a + 1, v.y);
    atomicAdd(a + 2, v.z); atomicAdd(a + 3, v.w);
    if (c == 0) atomicAdd(cnt + d, 1.0f);
}

__global__ void k_mean_div(float* __restrict__ acc, const float* __restrict__ cnt, int N) {
    int t = blockIdx.x * 256 + threadIdx.x;
    if (t >= N * 32) return;
    int i = t >> 5, c = (t & 31) * 4;
    float inv = 1.0f / fmaxf(cnt[i], 1.0f);
    float4 v = *(float4*)(acc + (size_t)i * 128 + c);
    v.x *= inv; v.y *= inv; v.z *= inv; v.w *= inv;
    *(float4*)(acc + (size_t)i * 128 + c) = v;
}

// ---------------------------------------------------------------------------
// SAGE layer GEMM: pre = gelu(mean@Wn + x@Wr + b), also per-column sum/sumsq.
// Block = 16 nodes x 128 cols (8 waves x one 16x16 WMMA tile each).
// ---------------------------------------------------------------------------
__global__ __launch_bounds__(256)
void k_sage_gemm(const float* __restrict__ x, int ldx, const float* __restrict__ mean,
                 const _Float16* __restrict__ WnP, const _Float16* __restrict__ WrP,
                 const float* __restrict__ bias,
                 float* __restrict__ pre,
                 float* __restrict__ colsum, float* __restrict__ colsumsq, int N) {
    __shared__ __align__(16) _Float16 Am[16 * 128];
    __shared__ __align__(16) _Float16 Ax[16 * 128];
    __shared__ float ls[256];
    int tid = threadIdx.x;
    int rowbase = blockIdx.x * 16;
    ls[tid] = 0.0f;
    for (int e = tid; e < 16 * 128; e += 256) {
        int r = e >> 7, k = e & 127;
        int rr = rowbase + r; if (rr >= N) rr = N - 1;
        Am[e] = (_Float16)mean[(size_t)rr * 128 + k];
        Ax[e] = (_Float16)x[(size_t)rr * ldx + k];
    }
    __syncthreads();
    int lane = tid & 31, wave = tid >> 5;
    int colbase = wave * 16;
    v8f acc = {};
#pragma unroll
    for (int kc = 0; kc < 4; ++kc)
        acc = wmma16(load_a_lds(Am, 128, kc * 32, lane), load_b_packed(WnP, 128, kc, colbase, lane), acc);
#pragma unroll
    for (int kc = 0; kc < 4; ++kc)
        acc = wmma16(load_a_lds(Ax, 128, kc * 32, lane), load_b_packed(WrP, 128, kc, colbase, lane), acc);
    int col = colbase + (lane & 15);
    int g = (lane >> 4) & 1;
    float bs = bias[col];
    float s = 0.0f, s2 = 0.0f;
#pragma unroll
    for (int vi = 0; vi < 8; ++vi) {
        int row = rowbase + g * 8 + vi;
        float v = gelu_f(acc[vi] + bs);
        if (row < N) {
            pre[(size_t)row * 128 + col] = v;
            s += v; s2 += v * v;
        }
    }
    atomicAdd(&ls[col], s);
    atomicAdd(&ls[128 + col], s2);
    __syncthreads();
    if (tid < 128) atomicAdd(&colsum[tid], ls[tid]);
    else           atomicAdd(&colsumsq[tid - 128], ls[tid]);
}

// GraphNorm: scale/shift per column from sum & sumsq.
__global__ void k_norm_finalize(const float* __restrict__ colsum, const float* __restrict__ colsumsq,
                                const float* __restrict__ gamma, const float* __restrict__ beta,
                                const float* __restrict__ alpha,
                                float* __restrict__ scale, float* __restrict__ shift, int N) {
    int t = threadIdx.x;                       // 128 threads
    float invN = 1.0f / (float)N;
    float m = colsum[t] * invN;
    float a = alpha[t];
    // E[(x-am)^2] = E[x^2] - 2am*E[x] + a^2 m^2,  E[x] = m
    float var = colsumsq[t] * invN - 2.0f * a * m * m + a * a * m * m;
    float sc = gamma[t] * rsqrtf(var + 1e-5f);
    scale[t] = sc;
    shift[t] = beta[t] - sc * a * m;
}

__global__ void k_norm_apply(const float* __restrict__ pre, const float* __restrict__ prev, int ldprev,
                             const float* __restrict__ scale, const float* __restrict__ shift,
                             float* __restrict__ out, int ldo, int N) {
    int t = blockIdx.x * 256 + threadIdx.x;
    if (t >= N * 32) return;
    int i = t >> 5, c = (t & 31) * 4;
    float4 v  = *(const float4*)(pre + (size_t)i * 128 + c);
    float4 s  = *(const float4*)(scale + c);
    float4 sh = *(const float4*)(shift + c);
    v.x = v.x * s.x + sh.x; v.y = v.y * s.y + sh.y;
    v.z = v.z * s.z + sh.z; v.w = v.w * s.w + sh.w;
    if (prev) {
        float4 p = *(const float4*)(prev + (size_t)i * ldprev + c);
        v.x += p.x; v.y += p.y; v.z += p.z; v.w += p.w;
    }
    *(float4*)(out + (size_t)i * ldo + c) = v;
}

// ---------------------------------------------------------------------------
// Generic MLP GEMM: out = act(A@W + b) [+resid], optional DyT on input.
// Block = 16 rows x 128 cols; grid.y covers Nout/128.
// ---------------------------------------------------------------------------
__global__ __launch_bounds__(256)
void k_mlp_gemm(const float* __restrict__ A, int lda, int K,
                const float* __restrict__ dytw, const float* __restrict__ dytb,
                const float* __restrict__ dyta, int use_dyt,
                const _Float16* __restrict__ Wp, int Nout, const float* __restrict__ bias,
                int act_gelu, const float* __restrict__ resid, int ldres,
                float* __restrict__ out, int ldo, int N) {
    __shared__ __align__(16) _Float16 As[16 * 384];
    int tid = threadIdx.x;
    int rowbase = blockIdx.x * 16;
    int colblock = blockIdx.y * 128;
    float alpha = use_dyt ? dyta[0] : 0.0f;
    for (int e = tid; e < (K << 4); e += 256) {
        int r = e / K, k = e - r * K;
        int rr = rowbase + r; if (rr >= N) rr = N - 1;
        float v = A[(size_t)rr * lda + k];
        if (use_dyt) v = dytw[k] * tanhf(alpha * v) + dytb[k];
        As[e] = (_Float16)v;
    }
    __syncthreads();
    int lane = tid & 31, wave = tid >> 5;
    int colbase = colblock + wave * 16;
    v8f acc = {};
    for (int kc = 0; kc < (K >> 5); ++kc)
        acc = wmma16(load_a_lds(As, K, kc * 32, lane), load_b_packed(Wp, Nout, kc, colbase, lane), acc);
    int col = colbase + (lane & 15);
    int g = (lane >> 4) & 1;
    float bs = bias[col];
#pragma unroll
    for (int vi = 0; vi < 8; ++vi) {
        int row = rowbase + g * 8 + vi;
        if (row >= N) continue;
        float v = acc[vi] + bs;
        if (act_gelu) v = gelu_f(v);
        if (resid) v += resid[(size_t)row * ldres + col];
        out[(size_t)row * ldo + col] = v;
    }
}

// Row L2-normalize z in place (one wave per node) + accumulate column sums.
__global__ __launch_bounds__(256)
void k_znorm(float* __restrict__ z, float* __restrict__ zsum, int N) {
    __shared__ float ls[128];
    int tid = threadIdx.x;
    if (tid < 128) ls[tid] = 0.0f;
    __syncthreads();
    int node = blockIdx.x * 8 + (tid >> 5);
    int lane = tid & 31;
    if (node < N) {
        float4 v = *(float4*)(z + (size_t)node * 128 + lane * 4);
        float ss = v.x * v.x + v.y * v.y + v.z * v.z + v.w * v.w;
        for (int o = 16; o > 0; o >>= 1) ss += __shfl_xor(ss, o, 32);
        float inv = 1.0f / (sqrtf(ss) + 1e-10f);
        v.x *= inv; v.y *= inv; v.z *= inv; v.w *= inv;
        *(float4*)(z + (size_t)node * 128 + lane * 4) = v;
        atomicAdd(&ls[lane * 4 + 0], v.x);
        atomicAdd(&ls[lane * 4 + 1], v.y);
        atomicAdd(&ls[lane * 4 + 2], v.z);
        atomicAdd(&ls[lane * 4 + 3], v.w);
    }
    __syncthreads();
    if (tid < 128) atomicAdd(&zsum[tid], ls[tid]);
}

// Jaccard head: jv = normalize(gelu(mean(z)@Wj1+bj1)@Wj2+bj2).  Single block.
__global__ void k_jaccard(const float* __restrict__ zsum,
                          const float* __restrict__ Wj1, const float* __restrict__ bj1,
                          const float* __restrict__ Wj2, const float* __restrict__ bj2,
                          float* __restrict__ jv, float invN) {
    __shared__ float jm[128], hh[128], red[128];
    int t = threadIdx.x;                       // 128 threads
    jm[t] = zsum[t] * invN;
    __syncthreads();
    float a = bj1[t];
    for (int k = 0; k < 128; ++k) a += jm[k] * Wj1[k * 128 + t];
    hh[t] = gelu_f(a);
    __syncthreads();
    float o = bj2[t];
    for (int k = 0; k < 128; ++k) o += hh[k] * Wj2[k * 128 + t];
    red[t] = o * o;
    __syncthreads();
    for (int s = 64; s > 0; s >>= 1) { if (t < s) red[t] += red[t + s]; __syncthreads(); }
    jv[t] = o / (sqrtf(red[0]) + 1e-10f);
}

// ---------------------------------------------------------------------------
// Fused contact MLP: 16 pairs per block; zc/h1/h2 stay in LDS (20 KB).
//   h1 = gelu(zc@W1+b1)  (256->256, 8 waves x 2 tiles)
//   h2 = gelu(h1@W2+b2)  (256->128, 8 waves x 1 tile)
//   p  = sigmoid(h2.Wc3 + bc3)
// ---------------------------------------------------------------------------
__global__ __launch_bounds__(256)
void k_contact(const float* __restrict__ z,
               const int* __restrict__ ia, const int* __restrict__ ib,
               const _Float16* __restrict__ W1p, const float* __restrict__ b1,
               const _Float16* __restrict__ W2p, const float* __restrict__ b2,
               const float* __restrict__ W3, const float* __restrict__ b3,
               float* __restrict__ probs, int EC) {
    __shared__ __align__(16) _Float16 Zc[16 * 256];
    __shared__ __align__(16) _Float16 H1[16 * 256];
    __shared__ __align__(16) _Float16 H2[16 * 128];
    int tid = threadIdx.x;
    int pb = blockIdx.x * 16;
    for (int e = tid; e < 16 * 256; e += 256) {
        int r = e >> 8, k = e & 255;
        int p = pb + r; if (p >= EC) p = EC - 1;
        int node = (k < 128) ? ia[p] : ib[p];
        Zc[e] = (_Float16)z[(size_t)node * 128 + (k & 127)];
    }
    __syncthreads();
    int lane = tid & 31, wave = tid >> 5;
    int g = (lane >> 4) & 1;
#pragma unroll
    for (int half = 0; half < 2; ++half) {
        int colbase = (wave + half * 8) * 16;
        v8f acc = {};
#pragma unroll
        for (int kc = 0; kc < 8; ++kc)
            acc = wmma16(load_a_lds(Zc, 256, kc * 32, lane), load_b_packed(W1p, 256, kc, colbase, lane), acc);
        int col = colbase + (lane & 15);
        float bs = b1[col];
#pragma unroll
        for (int vi = 0; vi < 8; ++vi) {
            int row = g * 8 + vi;
            H1[row * 256 + col] = (_Float16)gelu_f(acc[vi] + bs);
        }
    }
    __syncthreads();
    {
        int colbase = wave * 16;
        v8f acc = {};
#pragma unroll
        for (int kc = 0; kc < 8; ++kc)
            acc = wmma16(load_a_lds(H1, 256, kc * 32, lane), load_b_packed(W2p, 128, kc, colbase, lane), acc);
        int col = colbase + (lane & 15);
        float bs = b2[col];
#pragma unroll
        for (int vi = 0; vi < 8; ++vi) {
            int row = g * 8 + vi;
            H2[row * 128 + col] = (_Float16)gelu_f(acc[vi] + bs);
        }
    }
    __syncthreads();
    if (tid < 16) {
        int p = pb + tid;
        if (p < EC) {
            float s = b3[0];
            for (int k = 0; k < 128; ++k) s += (float)H2[tid * 128 + k] * W3[k];
            probs[p] = 1.0f / (1.0f + expf(-s));
        }
    }
}

// ---------------------------------------------------------------------------
// Host orchestration
// ---------------------------------------------------------------------------
extern "C" void kernel_launch(void* const* d_in, const int* in_sizes, int n_in,
                              void* d_out, int out_size, void* d_ws, size_t ws_size,
                              hipStream_t stream) {
    (void)n_in; (void)out_size; (void)ws_size;
    const int D  = 128;
    const int N  = in_sizes[0] / D;
    const int E  = in_sizes[1] / 2;
    const int EC = in_sizes[2] / 2;

    const float* x_res = (const float*)d_in[0];
    const int* src = (const int*)d_in[1];
    const int* dstn = src + E;
    const int* ia = (const int*)d_in[2];
    const int* ib = ia + EC;

    // params: flattened in setup_inputs() insertion order
    const float *Wn[3], *bn[3], *Wr[3], *gg[3], *gb[3], *ga[3];
    int idx = 3;
    for (int l = 0; l < 3; ++l) {
        Wn[l] = (const float*)d_in[idx++]; bn[l] = (const float*)d_in[idx++];
        Wr[l] = (const float*)d_in[idx++]; gg[l] = (const float*)d_in[idx++];
        gb[l] = (const float*)d_in[idx++]; ga[l] = (const float*)d_in[idx++];
    }
    const float* dyt_alpha = (const float*)d_in[idx++];   // 21 (scalar, read on device)
    const float* dyt_w = (const float*)d_in[idx++];       // 22
    const float* dyt_b = (const float*)d_in[idx++];       // 23
    const float* W1  = (const float*)d_in[idx++];         // 24 (384x256)
    const float* b1l = (const float*)d_in[idx++];
    const float* W2  = (const float*)d_in[idx++];         // 26 (256x256)
    const float* b2l = (const float*)d_in[idx++];
    const float* W3l = (const float*)d_in[idx++];         // 28 (256x128)
    const float* b3l = (const float*)d_in[idx++];
    const float* Wc1 = (const float*)d_in[idx++];         // 30 (256x256)
    const float* bc1 = (const float*)d_in[idx++];
    const float* Wc2 = (const float*)d_in[idx++];         // 32 (256x128)
    const float* bc2 = (const float*)d_in[idx++];
    const float* Wc3 = (const float*)d_in[idx++];         // 34 (128x1)
    const float* bc3 = (const float*)d_in[idx++];
    const float* Wj1 = (const float*)d_in[idx++];         // 36 (128x128)
    const float* bj1 = (const float*)d_in[idx++];
    const float* Wj2 = (const float*)d_in[idx++];         // 38 (128x128)
    const float* bj2 = (const float*)d_in[idx++];

    // workspace carve
    float* acc  = (float*)d_ws;                 // N*128
    float* pre  = acc + (size_t)N * D;          // N*128
    float* cnt  = pre + (size_t)N * D;          // N
    float* hcat = cnt + N;                      // N*384 (h2 aliases this after layers)
    float* h1   = hcat + (size_t)N * 384;       // N*256
    float* stats = h1 + (size_t)N * 256;        // 640 floats
    float* colsum   = stats;
    float* colsumsq = stats + 128;
    float* nscale   = stats + 256;
    float* nshift   = stats + 384;
    float* zsum     = stats + 512;
    _Float16* wp = (_Float16*)(stats + 640);
    _Float16 *WnP[3], *WrP[3];
    size_t o = 0;
    for (int l = 0; l < 3; ++l) { WnP[l] = wp + o; o += 16384; WrP[l] = wp + o; o += 16384; }
    _Float16* lin1p = wp + o; o += (size_t)384 * 256;
    _Float16* lin2p = wp + o; o += (size_t)256 * 256;
    _Float16* lin3p = wp + o; o += (size_t)256 * 128;
    _Float16* c1p   = wp + o; o += (size_t)256 * 256;
    _Float16* c2p   = wp + o; o += (size_t)256 * 128;

    float* outp  = (float*)d_out;
    float* jv    = outp;
    float* z     = outp + 128;
    float* probs = z + (size_t)N * D;

    auto gridFor = [](long long n) { return dim3((unsigned)((n + 255) / 256)); };

    // zero stats region (colsum/sumsq re-zeroed per layer; zsum once per call)
    k_fill_zero<<<gridFor(640), 256, 0, stream>>>(stats, 640);

    // pack all GEMM weights into WMMA-B f16 layout
    auto pack = [&](const float* W, _Float16* P, int K, int Nout) {
        k_pack_b<<<gridFor((long long)K * Nout), 256, 0, stream>>>(W, P, K, Nout);
    };
    for (int l = 0; l < 3; ++l) { pack(Wn[l], WnP[l], 128, 128); pack(Wr[l], WrP[l], 128, 128); }
    pack(W1, lin1p, 384, 256); pack(W2, lin2p, 256, 256); pack(W3l, lin3p, 256, 128);
    pack(Wc1, c1p, 256, 256);  pack(Wc2, c2p, 256, 128);

    const int mblocks = (N + 15) / 16;

    for (int l = 0; l < 3; ++l) {
        const float* xin = (l == 0) ? x_res : (hcat + (size_t)(l - 1) * D);
        int ldx = (l == 0) ? D : 384;
        k_fill_zero<<<gridFor((long long)N * D), 256, 0, stream>>>(acc, N * D);
        k_fill_zero<<<gridFor(N), 256, 0, stream>>>(cnt, N);
        k_fill_zero<<<gridFor(256), 256, 0, stream>>>(colsum, 256);
        k_scatter<<<gridFor((long long)E * 32), 256, 0, stream>>>(xin, ldx, src, dstn, acc, cnt, E);
        k_mean_div<<<gridFor((long long)N * 32), 256, 0, stream>>>(acc, cnt, N);
        k_sage_gemm<<<dim3(mblocks), 256, 0, stream>>>(xin, ldx, acc, WnP[l], WrP[l], bn[l],
                                                       pre, colsum, colsumsq, N);
        k_norm_finalize<<<1, 128, 0, stream>>>(colsum, colsumsq, gg[l], gb[l], ga[l], nscale, nshift, N);
        const float* prev = (l > 0) ? (hcat + (size_t)(l - 1) * D) : nullptr;
        k_norm_apply<<<gridFor((long long)N * 32), 256, 0, stream>>>(pre, prev, 384, nscale, nshift,
                                                                     hcat + (size_t)l * D, 384, N);
    }

    // Node MLP: DyT fused into GEMM1 input stage; h2 written over dead hcat.
    k_mlp_gemm<<<dim3(mblocks, 2), 256, 0, stream>>>(hcat, 384, 384, dyt_w, dyt_b, dyt_alpha, 1,
                                                     lin1p, 256, b1l, 1, nullptr, 0, h1, 256, N);
    k_mlp_gemm<<<dim3(mblocks, 2), 256, 0, stream>>>(h1, 256, 256, nullptr, nullptr, nullptr, 0,
                                                     lin2p, 256, b2l, 1, nullptr, 0, hcat, 256, N);
    k_mlp_gemm<<<dim3(mblocks, 1), 256, 0, stream>>>(hcat, 256, 256, nullptr, nullptr, nullptr, 0,
                                                     lin3p, 128, b3l, 0, x_res, 128, z, 128, N);

    k_znorm<<<dim3((N + 7) / 8), 256, 0, stream>>>(z, zsum, N);
    k_jaccard<<<1, 128, 0, stream>>>(zsum, Wj1, bj1, Wj2, bj2, jv, 1.0f / (float)N);
    k_contact<<<dim3((EC + 15) / 16), 256, 0, stream>>>(z, ia, ib, c1p, bc1, c2p, bc2, Wc3, bc3, probs, EC);
}
